// ArcFaceLoss_11802570130300
// MI455X (gfx1250) — matrix operand order repaired and checked by reading, not compiled, not run on hardware
//
#include <hip/hip_runtime.h>
#include <hip/hip_bf16.h>
#include <math.h>

typedef __attribute__((ext_vector_type(16))) _Float16 v16h;
typedef __attribute__((ext_vector_type(8)))  _Float16 v8h;
typedef __attribute__((ext_vector_type(8)))  float    v8f;
typedef __attribute__((ext_vector_type(4)))  float    v4f;

#define BATCH 8192
#define CLASSES 16384
#define DIM 256
#define BM 128
#define BN 128
#define NBLK (CLASSES / BN)     // 128 column-blocks per row
#define LDSP 264                // 256 + 8 f16 pad
#define CSP 132                 // 128 + 4 f32 pad for logits tile
#define SCALE 30.0f
#define COS_M 0.95533648912560601964f   // cos(0.3)
#define SIN_M 0.29552020666133957511f   // sin(0.3)

// Stage tiles with GLOBAL_LOAD_ASYNC_TO_LDS_B128 (ASYNCcnt path).
// Flip to 0 to fall back to global_load_b128 + ds_store_b128 staging.
#define USE_ASYNC_LDS 1

// smem layout (bytes):
//   [0,      67584)  As: 128 x 264 f16      -> recycled as Cs: 128 x 132 f32
//   [67584, 135168)  Bs: 128 x 264 f16      -> recycled as hm[256], hs[256] f32
//   [135168,135680)  Lb: 128 int labels
#define SMEM_BYTES 135680
#define AS_BYTES   67584

// ---------------------------------------------------------------------------
// Phase 1/2: L2-normalize rows of a [rows, 256] f32 matrix, emit f16.
// ---------------------------------------------------------------------------
__global__ void normalize_rows_f16(const float* __restrict__ src,
                                   _Float16* __restrict__ dst) {
    const int row = blockIdx.x;
    const int t   = threadIdx.x;
    const float v = src[(size_t)row * DIM + t];
    float s = v * v;
    #pragma unroll
    for (int o = 16; o > 0; o >>= 1) s += __shfl_down(s, o, 32);
    __shared__ float partial[8];
    __shared__ float inv_s;
    const int wave = t >> 5, lane = t & 31;
    if (lane == 0) partial[wave] = s;
    __syncthreads();
    if (t == 0) {
        float tot = 0.f;
        #pragma unroll
        for (int i = 0; i < 8; ++i) tot += partial[i];
        inv_s = 1.0f / fmaxf(sqrtf(tot), 1e-12f);
    }
    __syncthreads();
    dst[(size_t)row * DIM + t] = (_Float16)(v * inv_s);
}

// ---------------------------------------------------------------------------
// Phase 3: fused cos-sim GEMM + ArcFace margin + per-block softmax partials.
// v_wmma_f32_16x16x32_f16; block tile 128x128, 8 wave32 waves, wave tile 32x64.
// Tile staging goes directly to LDS via global_load_async_to_lds_b128.
// ---------------------------------------------------------------------------
__global__ __launch_bounds__(256)
void gemm_arcface_logits(const _Float16* __restrict__ X,
                         const _Float16* __restrict__ W,
                         const int* __restrict__ label,
                         float* __restrict__ logits,
                         float* __restrict__ pmax,
                         float* __restrict__ psum) {
    __shared__ __align__(16) char smem[SMEM_BYTES];
    _Float16* As = (_Float16*)smem;                 // [BM][LDSP]
    _Float16* Bs = As + BM * LDSP;                  // [BN][LDSP]
    int*      Lb = (int*)(smem + 2 * BM * LDSP * 2);// [BM]

    const int t  = threadIdx.x;
    const int bm = blockIdx.y * BM;   // batch offset
    const int bn = blockIdx.x * BN;   // class offset

#if USE_ASYNC_LDS
    // 4096 16B chunks per matrix; 16 chunks per thread, async direct-to-LDS.
    // VDST = per-lane LDS byte address (low 32 bits of generic LDS pointer),
    // VADDR = per-lane 32-bit byte offset, SADDR = 64-bit base (GVS mode).
    #pragma unroll
    for (int i = 0; i < 16; ++i) {
        const int chunk = t + i * 256;
        const int r = chunk >> 5;
        const int c = (chunk & 31) << 3;
        unsigned lds_a = (unsigned)(uintptr_t)&As[r * LDSP + c];
        unsigned lds_b = (unsigned)(uintptr_t)&Bs[r * LDSP + c];
        unsigned off_a = ((unsigned)(bm + r) * DIM + (unsigned)c) * 2u;
        unsigned off_b = ((unsigned)(bn + r) * DIM + (unsigned)c) * 2u;
        asm volatile("global_load_async_to_lds_b128 %0, %1, %2"
                     :: "v"(lds_a), "v"(off_a), "s"(X) : "memory");
        asm volatile("global_load_async_to_lds_b128 %0, %1, %2"
                     :: "v"(lds_b), "v"(off_b), "s"(W) : "memory");
    }
    if (t < BM) Lb[t] = label[bm + t];
    asm volatile("s_wait_asynccnt 0" ::: "memory");
#else
    #pragma unroll
    for (int i = 0; i < 16; ++i) {
        const int chunk = t + i * 256;
        const int r = chunk >> 5;
        const int c = (chunk & 31) << 3;
        *(v8h*)&As[r * LDSP + c] = *(const v8h*)&X[(size_t)(bm + r) * DIM + c];
        *(v8h*)&Bs[r * LDSP + c] = *(const v8h*)&W[(size_t)(bn + r) * DIM + c];
    }
    if (t < BM) Lb[t] = label[bm + t];
#endif
    __syncthreads();

    const int wave = t >> 5;
    const int lane = t & 31;
    const int wm = (wave >> 1) * 32;      // 0,32,64,96
    const int wn = (wave & 1) * 64;       // 0,64
    const int ml = lane & 15;
    const int h  = lane >> 4;

    v8f zero = {};
    v8f acc[2][4];
    #pragma unroll
    for (int i = 0; i < 2; ++i)
        #pragma unroll
        for (int j = 0; j < 4; ++j) acc[i][j] = zero;

    for (int ks = 0; ks < DIM; ks += 32) {
        v16h a[2];
        #pragma unroll
        for (int i = 0; i < 2; ++i) {
            const _Float16* p = &As[(wm + i * 16 + ml) * LDSP + ks + 8 * h];
            v8h lo = *(const v8h*)p;
            v8h hi = *(const v8h*)(p + 16);
            a[i] = __builtin_shufflevector(lo, hi, 0,1,2,3,4,5,6,7,
                                                   8,9,10,11,12,13,14,15);
        }
        v16h b[4];
        #pragma unroll
        for (int j = 0; j < 4; ++j) {
            const _Float16* p = &Bs[(wn + j * 16 + ml) * LDSP + ks + 8 * h];
            v8h lo = *(const v8h*)p;
            v8h hi = *(const v8h*)(p + 16);
            b[j] = __builtin_shufflevector(lo, hi, 0,1,2,3,4,5,6,7,
                                                   8,9,10,11,12,13,14,15);
        }
        #pragma unroll
        for (int i = 0; i < 2; ++i)
            #pragma unroll
            for (int j = 0; j < 4; ++j)
                acc[i][j] = __builtin_amdgcn_wmma_f32_16x16x32_f16(
                    false, a[i], false, b[j], (short)0, acc[i][j], false, false);
    }

    // All waves done reading As/Bs -> recycle LDS for the f32 logits tile.
    __syncthreads();
    float* Cs = (float*)smem;                       // [BM][CSP]
    float* hm = (float*)(smem + AS_BYTES);          // [BM][2] half-row maxes
    float* hs = (float*)(smem + AS_BYTES + 1024);   // [BM][2] half-row sums

    // Epilogue: clip, margin at label column, scale; store global + LDS tile.
    #pragma unroll
    for (int i = 0; i < 2; ++i) {
        #pragma unroll
        for (int j = 0; j < 4; ++j) {
            const int cl  = wn + j * 16 + ml;       // local col 0..127
            const int col = bn + cl;
            #pragma unroll
            for (int e = 0; e < 8; ++e) {
                const int rl  = wm + i * 16 + e + 8 * h;
                const int row = bm + rl;
                float c = acc[i][j][e];
                c = fminf(1.0f, fmaxf(-1.0f, c));
                float lg;
                if (Lb[rl] == col) {
                    const float sn =
                        __builtin_amdgcn_sqrtf(fmaxf(0.0f, 1.0f - c * c));
                    lg = SCALE * (c * COS_M - sn * SIN_M);
                } else {
                    lg = SCALE * c;
                }
                logits[(size_t)row * CLASSES + col] = lg;
                Cs[rl * CSP + cl] = lg;
            }
        }
    }
    __syncthreads();

    // Deterministic per-row (128 cols) max & sum-exp: 2 threads per row.
    const int r    = t >> 1;
    const int half = t & 1;
    const v4f* rp = (const v4f*)(Cs + r * CSP + half * 64);
    float m = -3.0e38f;
    #pragma unroll
    for (int k = 0; k < 16; ++k) {
        v4f v = rp[k];
        m = fmaxf(m, fmaxf(fmaxf(v.x, v.y), fmaxf(v.z, v.w)));
    }
    hm[r * 2 + half] = m;
    __syncthreads();
    const float gm = fmaxf(hm[r * 2], hm[r * 2 + 1]);
    float s = 0.f;
    #pragma unroll
    for (int k = 0; k < 16; ++k) {
        v4f v = rp[k];
        s += __expf(v.x - gm) + __expf(v.y - gm) +
             __expf(v.z - gm) + __expf(v.w - gm);
    }
    hs[r * 2 + half] = s;
    __syncthreads();
    if (half == 0) {
        pmax[(size_t)(bm + r) * NBLK + blockIdx.x] = gm;
        psum[(size_t)(bm + r) * NBLK + blockIdx.x] = hs[r * 2] + hs[r * 2 + 1];
    }
}

// ---------------------------------------------------------------------------
// Phase 4: merge 128 per-block partials per row -> NLL (128 threads / row).
// ---------------------------------------------------------------------------
__global__ void combine_nll(const float* __restrict__ pmax,
                            const float* __restrict__ psum,
                            const float* __restrict__ logits,
                            const int* __restrict__ label,
                            float* __restrict__ nll) {
    const int b = blockIdx.x;
    const int t = threadIdx.x;          // 0..127
    const int wave = t >> 5, lane = t & 31;
    __shared__ float pr[4];
    __shared__ float gmax_s, gsum_s;

    const float pm = pmax[(size_t)b * NBLK + t];
    const float ps = psum[(size_t)b * NBLK + t];

    float m = pm;
    #pragma unroll
    for (int o = 16; o > 0; o >>= 1) m = fmaxf(m, __shfl_xor(m, o, 32));
    if (lane == 0) pr[wave] = m;
    __syncthreads();
    if (t == 0) {
        float g = pr[0];
        #pragma unroll
        for (int i = 1; i < 4; ++i) g = fmaxf(g, pr[i]);
        gmax_s = g;
    }
    __syncthreads();
    const float g = gmax_s;

    float s = ps * __expf(pm - g);
    #pragma unroll
    for (int o = 16; o > 0; o >>= 1) s += __shfl_xor(s, o, 32);
    if (lane == 0) pr[wave] = s;
    __syncthreads();
    if (t == 0) {
        float tot = 0.f;
        #pragma unroll
        for (int i = 0; i < 4; ++i) tot += pr[i];
        gsum_s = tot;
        const float tgt = logits[(size_t)b * CLASSES + label[b]];
        nll[b] = (g + __logf(gsum_s)) - tgt;
    }
}

// ---------------------------------------------------------------------------
// Phase 5: mean of per-row NLL -> loss scalar.
// ---------------------------------------------------------------------------
__global__ void reduce_loss(const float* __restrict__ nll, float* __restrict__ out) {
    const int t = threadIdx.x;
    float s = 0.f;
    for (int i = t; i < BATCH; i += 256) s += nll[i];
    #pragma unroll
    for (int o = 16; o > 0; o >>= 1) s += __shfl_down(s, o, 32);
    __shared__ float pr[8];
    const int wave = t >> 5, lane = t & 31;
    if (lane == 0) pr[wave] = s;
    __syncthreads();
    if (t == 0) {
        float tot = 0.f;
        #pragma unroll
        for (int i = 0; i < 8; ++i) tot += pr[i];
        out[0] = tot / (float)BATCH;
    }
}

extern "C" void kernel_launch(void* const* d_in, const int* in_sizes, int n_in,
                              void* d_out, int out_size, void* d_ws, size_t ws_size,
                              hipStream_t stream) {
    (void)in_sizes; (void)n_in; (void)out_size; (void)ws_size;
    const float* inp    = (const float*)d_in[0];   // [8192, 256]
    const int*   label  = (const int*)d_in[1];     // [8192]
    const float* weight = (const float*)d_in[2];   // [16384, 256]

    float* out    = (float*)d_out;                 // out[0]=loss, out+1 = logits
    float* logits = out + 1;

    // Workspace: X_f16 (4MB) | W_f16 (8MB) | pmax (4MB) | psum (4MB) | nll (32KB)
    _Float16* xf   = (_Float16*)d_ws;
    _Float16* wf   = xf + (size_t)BATCH * DIM;
    float*    pmax = (float*)(wf + (size_t)CLASSES * DIM);
    float*    psum = pmax + (size_t)BATCH * NBLK;
    float*    nll  = psum + (size_t)BATCH * NBLK;

    normalize_rows_f16<<<BATCH,   256, 0, stream>>>(inp,    xf);
    normalize_rows_f16<<<CLASSES, 256, 0, stream>>>(weight, wf);

    dim3 grid(CLASSES / BN, BATCH / BM);   // (128, 64)
    gemm_arcface_logits<<<grid, 256, 0, stream>>>(xf, wf, label, logits,
                                                  pmax, psum);

    combine_nll<<<BATCH, 128, 0, stream>>>(pmax, psum, logits, label, nll);
    reduce_loss<<<1, 256, 0, stream>>>(nll, out);
}
